// DeformConv1d_90237262889597
// MI455X (gfx1250) — compile-verified, matrix operand order checked by reading, not run on hardware
//
#include <hip/hip_runtime.h>
#include <hip/hip_bf16.h>

#define Bn     8
#define Cn     256
#define Ln     4096
#define COUTn  256
#define Kn     5
#define Gn     4
#define Dn     2
#define CPGn   32      // C / (G*D)
#define KOFFn  7
#define PADOFFn 3
#define PADn   2
#define Pn     4096    // output length (stride=1, dil=1)
#define KTOT   320     // D*CPG*K contraction length
#define NT     64      // positions per block in main GEMM kernel
#define KP     (CPGn * KOFFn)   // 224 = predictor contraction length
#define TPB    128              // positions per block in predictor kernel

typedef __attribute__((ext_vector_type(16))) __bf16 v16bf;
typedef __attribute__((ext_vector_type(2)))  __bf16 v2bf;
typedef __attribute__((ext_vector_type(8)))  float  v8f;

// ---- Kernel P0: pack predictor weights (off rows 0-4, mask rows 5-9, pad 16)
// pws[(gd*7 + s)*32 + lane] : bf16 A fragment, K = (c*7 + t), 7 steps of 32
__global__ __launch_bounds__(224) void pack_pred_weights_kernel(
    const float* __restrict__ w_off, const float* __restrict__ w_mask,
    v16bf* __restrict__ pws)
{
    int gd   = blockIdx.x;            // 0..7
    int s    = threadIdx.x >> 5;      // 0..6
    int lane = threadIdx.x & 31;
    int m    = lane & 15;
    const float* src = nullptr;
    if (m < 5)        src = w_off  + (size_t)(gd * Kn + m) * KP;
    else if (m < 10)  src = w_mask + (size_t)(gd * Kn + (m - 5)) * KP;
    int khalf = (lane >= 16) ? 8 : 0;
    v16bf a;
    #pragma unroll
    for (int vg = 0; vg < 8; ++vg) {
        int koff = ((vg >= 4) ? 16 : 0) + khalf + (vg & 3) * 2;  // pairs contiguous
        float2 w2 = src ? *reinterpret_cast<const float2*>(src + s * 32 + koff)
                        : make_float2(0.0f, 0.0f);
        a[2 * vg]     = (__bf16)w2.x;
        a[2 * vg + 1] = (__bf16)w2.y;
    }
    pws[(size_t)(gd * 7 + s) * 32 + lane] = a;
}

// ---- Kernel P1: pack main GEMM A fragments (bf16, WMMA register layout) ----
// aws[((g*4 + m_tile)*10 + s)*32 + lane]
__global__ __launch_bounds__(320) void pack_weights_kernel(
    const float* __restrict__ weight, v16bf* __restrict__ aws)
{
    int gm   = blockIdx.x;            // g*4 + m_tile
    int g    = gm >> 2, m_tile = gm & 3;
    int s    = threadIdx.x >> 5;      // 0..9
    int lane = threadIdx.x & 31;
    int m_row = m_tile * 16 + (lane & 15);
    const float* wrow = weight + (size_t)(g * 64 + m_row) * KTOT;
    int khalf = (lane >= 16) ? 8 : 0;
    v16bf a;
    #pragma unroll
    for (int vg = 0; vg < 8; ++vg) {
        int koff = ((vg >= 4) ? 16 : 0) + khalf + (vg & 3) * 2;
        float2 w2 = *reinterpret_cast<const float2*>(wrow + s * 32 + koff);
        a[2 * vg]     = (__bf16)w2.x;
        a[2 * vg + 1] = (__bf16)w2.y;
    }
    aws[(size_t)(gm * 10 + s) * 32 + lane] = a;
}

// ---------------- Kernel 1: predictor convs as WMMA + softmax epilogue -------
__global__ __launch_bounds__(256) void predict_wmma_kernel(
    const float* __restrict__ x,     const v16bf* __restrict__ pws,
    const float* __restrict__ b_off, const float* __restrict__ b_mask,
    float* __restrict__ pos_ws,      float* __restrict__ msk_ws)
{
    const int XW2 = TPB + KOFFn - 1;                     // 134
    __shared__ __align__(32) unsigned bfr[KP * TPB / 2]; // 56 KB, fragment layout
    __shared__ float xs[CPGn][XW2];                      // ~17 KB
    __shared__ float sout[16][TPB];                      // 8 KB

    int blk = blockIdx.x;
    int pt  = blk % (Pn / TPB);
    int gd  = (blk / (Pn / TPB)) % (Gn * Dn);
    int b   = blk / ((Pn / TPB) * (Gn * Dn));
    int tid  = threadIdx.x;
    int w    = tid >> 5;
    int lane = tid & 31;
    int p0   = pt * TPB;

    // stage x slab [32 x 134] with zero padding
    const float* xb = x + ((size_t)b * Cn + (size_t)gd * CPGn) * Ln;
    for (int i = tid; i < CPGn * XW2; i += 256) {
        int c = i / XW2, col = i % XW2;
        int gp = p0 - PADOFFn + col;
        xs[c][col] = (gp >= 0 && gp < Ln) ? xb[(size_t)c * Ln + gp] : 0.0f;
    }

    // A fragments: coalesced global_load_b128 pairs
    const v16bf* abase = pws + (size_t)(gd * 7) * 32 + lane;
    v16bf afrag[7];
    #pragma unroll
    for (int s = 0; s < 7; ++s) afrag[s] = abase[s * 32];

    __syncthreads();

    // build im2col B tile in WMMA fragment layout (pair-packed dword stores)
    {
        int pl = tid & 127;
        int r  = (tid >> 7) * 2;     // this thread covers even rows r, r+4, ...
        int c  = 0, t = r;           // r = c*7 + t
        #pragma unroll 4
        for (int it = 0; it < 56; ++it) {
            float v0 = xs[c][pl + t];
            int c1 = c, t1 = t + 1;
            if (t1 == KOFFn) { t1 = 0; c1++; }
            float v1 = xs[c1][pl + t1];
            int kk = r & 31, s = r >> 5;
            int hi = (kk >> 3) & 1;
            int vg = ((kk >> 4) & 1) * 4 + ((kk >> 1) & 3);
            int nt = pl >> 4, n = pl & 15;
            union { v2bf h; unsigned u; } pk;
            pk.h[0] = (__bf16)v0;
            pk.h[1] = (__bf16)v1;
            bfr[((s * 8 + nt) * 32 + (n + 16 * hi)) * 8 + vg] = pk.u;
            r += 4; t += 4;
            if (t >= KOFFn) { t -= KOFFn; c++; }
        }
    }
    __syncthreads();

    // WMMA: wave w -> 16-column tile, K = 224 in 7 steps
    {
        const v16bf* vsf = reinterpret_cast<const v16bf*>(bfr);
        v8f acc = {};
        #pragma unroll
        for (int s = 0; s < 7; ++s) {
            v16bf bfrag = vsf[(s * 8 + w) * 32 + lane];
            acc = __builtin_amdgcn_wmma_f32_16x16x32_bf16(
                false, afrag[s], false, bfrag, (short)0, acc, false, false);
        }
        int hi  = (lane >> 4) & 1;
        int col = w * 16 + (lane & 15);
        #pragma unroll
        for (int r = 0; r < 8; ++r) sout[hi * 8 + r][col] = acc[r];
    }
    __syncthreads();

    // epilogue: bias + softmax over taps + sampling positions
    if (tid < TPB) {
        int p = p0 + tid;
        float ao[Kn], am[Kn];
        #pragma unroll
        for (int k = 0; k < Kn; ++k) {
            ao[k] = sout[k][tid]     + b_off[gd * Kn + k];
            am[k] = sout[5 + k][tid] + b_mask[gd * Kn + k];
        }
        float mx = am[0];
        #pragma unroll
        for (int k = 1; k < Kn; ++k) mx = fmaxf(mx, am[k]);
        float e[Kn], ssum = 0.0f;
        #pragma unroll
        for (int k = 0; k < Kn; ++k) { e[k] = __expf(am[k] - mx); ssum += e[k]; }
        float inv = 1.0f / ssum;
        size_t base = (((size_t)b * (Gn * Dn) + gd) * Kn) * Pn + p;
        #pragma unroll
        for (int k = 0; k < Kn; ++k) {
            pos_ws[base + (size_t)k * Pn] = (float)(p - PADn + k) + ao[k];
            msk_ws[base + (size_t)k * Pn] = e[k] * inv;
        }
    }
}

// ---------------- Kernel 2: deformable gather + bf16 WMMA GEMM ---------------
__global__ __launch_bounds__(256) void deform_wmma_kernel(
    const float* __restrict__ x,      const v16bf* __restrict__ aws,
    const float* __restrict__ bias,   const float* __restrict__ pos_ws,
    const float* __restrict__ msk_ws, float* __restrict__ out)
{
    __shared__ __align__(32) unsigned vs32[KTOT * NT / 2];  // 40 KB, fragment layout
    __shared__ float4 sprm[Dn * Kn * NT];                   // 10 KB

    const int NPT = Pn / NT;
    int blk = blockIdx.x;
    int pt  = blk % NPT;
    int g   = (blk / NPT) % Gn;
    int b   = blk / (NPT * Gn);
    int tid  = threadIdx.x;
    int w    = tid >> 5;
    int lane = tid & 31;
    int p0   = pt * NT;

    int m_tile = w & 3;
    const v16bf* abase = aws + (size_t)((g * 4 + m_tile) * 10) * 32 + lane;
    v16bf afrag[10];
    #pragma unroll
    for (int s = 0; s < 10; ++s) afrag[s] = abase[s * 32];

    for (int e = tid; e < Dn * Kn * NT; e += 256) {
        int d   = e / (Kn * NT);
        int rem = e % (Kn * NT);
        int k   = rem / NT;
        int pl  = rem % NT;
        size_t idx = (((size_t)b * (Gn * Dn) + g * Dn + d) * Kn + k) * Pn + p0 + pl;
        float pos = pos_ws[idx];
        float m   = msk_ws[idx];
        float fl  = floorf(pos);
        int   i0  = (int)fl;
        int   i1  = i0 + 1;
        float fr  = pos - fl;
        float ok0 = (i0 >= 0 && i0 < Ln) ? 1.0f : 0.0f;
        float ok1 = (i1 >= 0 && i1 < Ln) ? 1.0f : 0.0f;
        float4 q;
        q.x = __int_as_float(min(max(i0, 0), Ln - 1));
        q.y = __int_as_float(min(max(i1, 0), Ln - 1));
        q.z = (1.0f - fr) * ok0 * m;
        q.w = fr * ok1 * m;
        sprm[e] = q;
    }
    __syncthreads();

    const float* xb = x + ((size_t)b * Cn + (size_t)g * Dn * CPGn) * Ln;
    for (int e = tid; e < Dn * (CPGn / 2) * NT; e += 256) {
        int d   = e / ((CPGn / 2) * NT);
        int rem = e % ((CPGn / 2) * NT);
        int c2  = rem / NT;
        int pl  = rem % NT;
        int c0  = c2 * 2;
        const float* xr0 = xb + ((size_t)d * CPGn + c0) * Ln;
        const float* xr1 = xr0 + Ln;
        float vals[10];
        #pragma unroll
        for (int k = 0; k < Kn; ++k) {
            float4 q = sprm[(d * Kn + k) * NT + pl];
            int i0 = __float_as_int(q.x);
            int i1 = __float_as_int(q.y);
            vals[k]     = xr0[i0] * q.z + xr0[i1] * q.w;
            vals[5 + k] = xr1[i0] * q.z + xr1[i1] * q.w;
        }
        int r0 = (d * CPGn + c0) * Kn;      // even
        int nt = pl >> 4, n = pl & 15;
        #pragma unroll
        for (int i = 0; i < 5; ++i) {
            int r  = r0 + 2 * i;
            int kk = r & 31, s = r >> 5;
            int hi = (kk >> 3) & 1;
            int vg = ((kk >> 4) & 1) * 4 + ((kk >> 1) & 3);
            int dlane = n + 16 * hi;
            union { v2bf h; unsigned u; } pk;
            pk.h[0] = (__bf16)vals[2 * i];
            pk.h[1] = (__bf16)vals[2 * i + 1];
            vs32[((s * 4 + nt) * 32 + dlane) * 8 + vg] = pk.u;
        }
    }
    __syncthreads();

    const v16bf* vsf = reinterpret_cast<const v16bf*>(vs32);
    int nt0 = (w >> 2) * 2;
    int hi  = (lane >> 4) & 1;
    #pragma unroll
    for (int t = 0; t < 2; ++t) {
        int nt = nt0 + t;
        v8f acc = {};
        #pragma unroll
        for (int s = 0; s < 10; ++s) {
            v16bf bfrag = vsf[(s * 4 + nt) * 32 + lane];
            acc = __builtin_amdgcn_wmma_f32_16x16x32_bf16(
                false, afrag[s], false, bfrag, (short)0, acc, false, false);
        }
        int pcol = p0 + nt * 16 + (lane & 15);
        #pragma unroll
        for (int r = 0; r < 8; ++r) {
            int oc = g * 64 + m_tile * 16 + hi * 8 + r;
            out[((size_t)b * COUTn + oc) * Pn + pcol] = acc[r] + bias[oc];
        }
    }
}

extern "C" void kernel_launch(void* const* d_in, const int* in_sizes, int n_in,
                              void* d_out, int out_size, void* d_ws, size_t ws_size,
                              hipStream_t stream) {
    (void)in_sizes; (void)n_in; (void)out_size; (void)ws_size;
    const float* x      = (const float*)d_in[0];
    const float* w_off  = (const float*)d_in[1];
    const float* b_off  = (const float*)d_in[2];
    const float* w_mask = (const float*)d_in[3];
    const float* b_mask = (const float*)d_in[4];
    const float* weight = (const float*)d_in[5];
    const float* bias   = (const float*)d_in[6];
    float* out = (float*)d_out;

    size_t nposmsk = (size_t)Bn * (Gn * Dn) * Kn * Pn;   // 1,310,720
    float* pos_ws = (float*)d_ws;
    float* msk_ws = pos_ws + nposmsk;
    v16bf* aws    = (v16bf*)(msk_ws + nposmsk);          // 160 KB
    v16bf* pws    = aws + (size_t)16 * 10 * 32;          // 56 KB

    pack_pred_weights_kernel<<<Gn * Dn, 224, 0, stream>>>(w_off, w_mask, pws);
    pack_weights_kernel<<<Gn * 4, 320, 0, stream>>>(weight, aws);
    predict_wmma_kernel<<<Bn * (Gn * Dn) * (Pn / TPB), 256, 0, stream>>>(
        x, pws, b_off, b_mask, pos_ws, msk_ws);
    deform_wmma_kernel<<<Bn * Gn * (Pn / NT), 256, 0, stream>>>(
        x, aws, bias, pos_ws, msk_ws, out);
}
